// upl_ea_13932873909155
// MI455X (gfx1250) — matrix operand order repaired and checked by reading, not compiled, not run on hardware
//
#include <hip/hip_runtime.h>
#include <math.h>

// ---------------------------------------------------------------------------
// Types for CDNA5 WMMA (wave32, 16x16x32 bf16 -> f32)
// ---------------------------------------------------------------------------
typedef __attribute__((ext_vector_type(16))) __bf16 v16bf;
typedef __attribute__((ext_vector_type(8)))  float  v8f;

struct __align__(16) Q4 { unsigned int x, y, z, w; };
struct U32x8 { Q4 lo, hi; };

__device__ __forceinline__ unsigned short f2bf(float f) {
    unsigned int u = __float_as_uint(f);
    u += 0x7FFFu + ((u >> 16) & 1u);   // round-to-nearest-even
    return (unsigned short)(u >> 16);
}

// ---------------------------------------------------------------------------
// Row L2-normalize: embn[r] = we[r] / ||we[r]||   (wave per row, D=300)
// ---------------------------------------------------------------------------
__global__ __launch_bounds__(256) void k_norm(const float* __restrict__ we,
                                              float* __restrict__ embn,
                                              int E, int D) {
    int g    = blockIdx.x * blockDim.x + threadIdx.x;
    int row  = g >> 5;
    int lane = g & 31;
    if (row >= E) return;
    const float* src = we + (size_t)row * D;
    float s = 0.f;
    for (int d = lane; d < D; d += 32) { float v = src[d]; s += v * v; }
    #pragma unroll
    for (int o = 16; o >= 1; o >>= 1) s += __shfl_xor(s, o, 32);
    float inv = 1.0f / sqrtf(s);
    float* dst = embn + (size_t)row * D;
    for (int d = lane; d < D; d += 32) dst[d] = src[d] * inv;
}

// ---------------------------------------------------------------------------
// COO SpMM (scatter form): out[rows[e], colOff + :] += vals[e] * x[cols[e], :]
// wave per nnz; coalesced lanes over the feature dim; f32 atomics into L2.
// ---------------------------------------------------------------------------
__global__ __launch_bounds__(256) void k_spmm(const int* __restrict__ rows,
                                              const int* __restrict__ cols,
                                              const float* __restrict__ vals,
                                              const float* __restrict__ x, int xld,
                                              float* __restrict__ out, int old_,
                                              int colOff, int width, int nnz) {
    int g    = blockIdx.x * blockDim.x + threadIdx.x;
    int e    = g >> 5;
    int lane = g & 31;
    if (e >= nnz) return;
    int r = rows[e], c = cols[e];
    float v = vals[e];
    const float* xp = x + (size_t)c * xld;
    float* op = out + (size_t)r * old_ + colOff;
    for (int d = lane; d < width; d += 32) atomicAdd(&op[d], v * xp[d]);
}

// r_emb rows [R, 2R) = -r_emb rows [0, R)
__global__ __launch_bounds__(256) void k_negmirror(float* __restrict__ remb, int n) {
    int g = blockIdx.x * blockDim.x + threadIdx.x;
    if (g >= n) return;
    remb[(size_t)n + g] = -remb[g];
}

// ---------------------------------------------------------------------------
// Tiled GEMM with fused epilogues, fp32 in, bf16 WMMA, fp32 accumulate.
//   C[M,N] = epilogue( A[M,K] @ B[K,N] ),  A = [A1 | A2]
//   OP 0: C = acc
//   OP 1: C = res1 + relu(acc + bias)                       (entity update)
//   OP 2: tg = sigmoid(acc + bias); C = tg*relu(res2) + (1-tg)*res1  (highway)
// Block: 256 thr = 8 waves (4 row-groups x 2 col-groups). Tile 128x160,
// K-step 32. Wave -> 32x80 subtile = 2x5 accumulators (10 v8f = 80 VGPRs).
// A re-read factor for N=300 drops to ceil(300/160)=2 (was 10 at TN=32),
// cutting GEMM HBM/L2 traffic ~5x — the dominant cost at 23.3 TB/s.
// LDS row stride 40 halves => conflict-free, 16B-aligned b128 fragment loads
// matching the CDNA5 16-bit A (16x32) and B (32x16) WMMA VGPR layouts.
// ---------------------------------------------------------------------------
template <int OP>
__global__ __launch_bounds__(256) void gemm_bf16(
    const float* __restrict__ A1, int w1,
    const float* __restrict__ A2,
    const float* __restrict__ B, int Kdim, int N, int M,
    const float* __restrict__ bias,
    const float* __restrict__ res1,
    const float* __restrict__ res2,
    float* __restrict__ C) {

    constexpr int TM = 128, TN = 160, TK = 32, LDA = 40, LDB = 40;
    constexpr int NJ = 5;                       // 5 n-subtiles of 16 per wave
    __shared__ __align__(16) unsigned short sA[TM * LDA];
    __shared__ __align__(16) unsigned short sB[TN * LDB];

    const int tid   = threadIdx.x;
    const int lane  = tid & 31;
    const int w     = tid >> 5;
    const int wr    = (w & 3) * 32;             // wave row base inside tile
    const int wc    = (w >> 2) * 80;            // wave col base inside tile
    const int mBase = blockIdx.y * TM;
    const int nBase = blockIdx.x * TN;
    const int w2    = Kdim - w1;

    const int mA0 = wr + (lane & 15);           // A-frag row (low 16 of subtile)
    const int kh  = (lane >> 4) * 8;            // A-frag K half-select
    const int nL  = wc + (lane & 15);           // B-frag column (local, j=0)
    const int kb  = (lane >> 4) * 8;            // B-frag K base

    v8f acc0[NJ], acc1[NJ];
    #pragma unroll
    for (int j = 0; j < NJ; ++j) { acc0[j] = {}; acc1[j] = {}; }

    const int kSteps = (Kdim + TK - 1) / TK;

    for (int kt = 0; kt < kSteps; ++kt) {
        const int kBase = kt * TK;
        // ---- stage A tile (128x32 fp32 -> bf16) ----
        for (int i = tid; i < TM * TK; i += 256) {
            int row = i >> 5, col = i & 31;
            int gm = mBase + row, gk = kBase + col;
            float v = 0.f;
            if (gm < M && gk < Kdim)
                v = (gk < w1) ? A1[(size_t)gm * w1 + gk]
                              : A2[(size_t)gm * w2 + (gk - w1)];
            sA[row * LDA + col] = f2bf(v);
            if (col == 0 && gm < M) {           // prefetch next K tile
                int gk2 = kBase + TK;
                if (gk2 < Kdim)
                    __builtin_prefetch((gk2 < w1)
                        ? (const void*)&A1[(size_t)gm * w1 + gk2]
                        : (const void*)&A2[(size_t)gm * w2 + (gk2 - w1)], 0, 1);
            }
        }
        // ---- stage B tile (32x160), transposed: sB[n][k] ----
        for (int i = tid; i < TK * TN; i += 256) {
            int kr = i / TN, nc = i % TN;       // consecutive i -> consecutive n
            int gk = kBase + kr, gn = nBase + nc;
            float v = (gk < Kdim && gn < N) ? B[(size_t)gk * N + gn] : 0.f;
            sB[nc * LDB + kr] = f2bf(v);
        }
        __syncthreads();

        // ---- A fragments (two 16-row groups) per documented CDNA5 layout ----
        U32x8 ua0, ua1;
        const Q4* p;
        p = (const Q4*)&sA[mA0 * LDA + kh];             ua0.lo = p[0];
        p = (const Q4*)&sA[mA0 * LDA + 16 + kh];        ua0.hi = p[0];
        p = (const Q4*)&sA[(mA0 + 16) * LDA + kh];      ua1.lo = p[0];
        p = (const Q4*)&sA[(mA0 + 16) * LDA + 16 + kh]; ua1.hi = p[0];
        v16bf a0 = __builtin_bit_cast(v16bf, ua0);
        v16bf a1 = __builtin_bit_cast(v16bf, ua1);

        #pragma unroll
        for (int j = 0; j < NJ; ++j) {
            U32x8 ub;
            const Q4* pb = (const Q4*)&sB[(nL + j * 16) * LDB + kb];
            ub.lo = pb[0]; ub.hi = pb[2];
            v16bf b = __builtin_bit_cast(v16bf, ub);
            acc0[j] = __builtin_amdgcn_wmma_f32_16x16x32_bf16(
                          false, a0, false, b, (short)0, acc0[j], false, false);
            acc1[j] = __builtin_amdgcn_wmma_f32_16x16x32_bf16(
                          false, a1, false, b, (short)0, acc1[j], false, false);
        }
        __syncthreads();
    }

    // ---- epilogue: C/D layout: element r -> m = base + 8*(lane>>4) + r ----
    #pragma unroll
    for (int j = 0; j < NJ; ++j) {
        int gn = nBase + nL + j * 16;
        if (gn >= N) continue;
        #pragma unroll
        for (int r = 0; r < 8; ++r) {
            int gm0 = mBase + wr + 8 * (lane >> 4) + r;
            int gm1 = gm0 + 16;
            float v0 = acc0[j][r], v1 = acc1[j][r];
            if (gm0 < M) {
                size_t o = (size_t)gm0 * N + gn;
                if (OP == 0) C[o] = v0;
                else if (OP == 1) C[o] = res1[o] + fmaxf(v0 + bias[gn], 0.f);
                else {
                    float tg = 1.f / (1.f + expf(-(v0 + bias[gn])));
                    C[o] = tg * fmaxf(res2[o], 0.f) + (1.f - tg) * res1[o];
                }
            }
            if (gm1 < M) {
                size_t o = (size_t)gm1 * N + gn;
                if (OP == 0) C[o] = v1;
                else if (OP == 1) C[o] = res1[o] + fmaxf(v1 + bias[gn], 0.f);
                else {
                    float tg = 1.f / (1.f + expf(-(v1 + bias[gn])));
                    C[o] = tg * fmaxf(res2[o], 0.f) + (1.f - tg) * res1[o];
                }
            }
        }
    }
}

// ---------------------------------------------------------------------------
// Hinge alignment loss. One 256-thread block per pair t.
// Waves split the K=25 negatives; lanes stride the D=300 dims.
// ---------------------------------------------------------------------------
__global__ __launch_bounds__(256) void k_loss(const float* __restrict__ node,
                                              const int* __restrict__ pl,
                                              const int* __restrict__ pr,
                                              const int* __restrict__ negR,
                                              const int* __restrict__ negL,
                                              const float* __restrict__ mask,
                                              float* __restrict__ out) {
    constexpr int D = 300, K = 25;
    __shared__ float lx[D], rx[D];
    __shared__ float red[8], wacc[8];
    __shared__ float sDm;

    const int t = blockIdx.x;
    const int tid = threadIdx.x, lane = tid & 31, w = tid >> 5;
    const int li = pl[t], ri = pr[t];

    float p = 0.f;
    for (int d = tid; d < D; d += 256) {
        float a = node[(size_t)li * D + d];
        float b = node[(size_t)ri * D + d];
        lx[d] = a; rx[d] = b; p += fabsf(a - b);
    }
    #pragma unroll
    for (int o = 16; o >= 1; o >>= 1) p += __shfl_xor(p, o, 32);
    if (lane == 0) red[w] = p;
    __syncthreads();
    if (tid == 0) {
        float s = 0.f;
        for (int i = 0; i < 8; ++i) s += red[i];
        sDm = s + 1.0f;                      // + GAMMA
    }
    __syncthreads();
    const float Dm = sDm;

    float acc = 0.f;
    for (int k = w; k < K; k += 8) {
        int ir = negR[t * K + k], il = negL[t * K + k];
        const float* nr = node + (size_t)ir * D;
        const float* nl = node + (size_t)il * D;
        float s1 = 0.f, s2 = 0.f;
        for (int d = lane; d < D; d += 32) {
            s1 += fabsf(lx[d] - nr[d]);
            s2 += fabsf(nl[d] - rx[d]);
        }
        #pragma unroll
        for (int o = 16; o >= 1; o >>= 1) {
            s1 += __shfl_xor(s1, o, 32);
            s2 += __shfl_xor(s2, o, 32);
        }
        if (lane == 0) {
            float mk = mask[t * K + k];
            acc += fmaxf(Dm - s1, 0.f) * mk + fmaxf(Dm - s2, 0.f) * mk;
        }
    }
    if (lane == 0) wacc[w] = acc;
    __syncthreads();
    if (tid == 0) {
        float s = 0.f;
        for (int i = 0; i < 8; ++i) s += wacc[i];
        atomicAdd(out, 0.5f * s);
    }
}

// ---------------------------------------------------------------------------
extern "C" void kernel_launch(void* const* d_in, const int* in_sizes, int n_in,
                              void* d_out, int out_size, void* d_ws, size_t ws_size,
                              hipStream_t stream) {
    const int D = 300, R = 1000;
    const int E = in_sizes[0] / D;

    const float* we   = (const float*)d_in[0];
    const float* kg   = (const float*)d_in[1];
    const float* bg   = (const float*)d_in[2];
    const float* W1   = (const float*)d_in[3];
    const float* W2   = (const float*)d_in[4];
    const float* Dw   = (const float*)d_in[5];
    const float* Bias = (const float*)d_in[6];
    const int*   hr_r = (const int*)d_in[7];
    const int*   hr_c = (const int*)d_in[8];
    const float* hr_v = (const float*)d_in[9];
    const int*   tr_r = (const int*)d_in[10];
    const int*   tr_c = (const int*)d_in[11];
    const float* tr_v = (const float*)d_in[12];
    const int*   er_r = (const int*)d_in[13];
    const int*   er_c = (const int*)d_in[14];
    const float* er_v = (const float*)d_in[15];
    const int*   ad_r = (const int*)d_in[16];
    const int*   ad_c = (const int*)d_in[17];
    const float* ad_v = (const float*)d_in[18];
    const int*   posL = (const int*)d_in[19];
    const int*   posR = (const int*)d_in[20];
    const int*   negR = (const int*)d_in[21];
    const int*   negL = (const int*)d_in[22];
    const float* mask = (const float*)d_in[23];

    const int nnzHR = in_sizes[7], nnzTR = in_sizes[10];
    const int nnzER = in_sizes[13], nnzAD = in_sizes[16];
    const int T = in_sizes[19];

    // Workspace layout (aliased across phases):
    char* ws = (char*)d_ws;
    const size_t SZ = (size_t)E * D * sizeof(float);      // one [E,D] f32 buffer
    float* embn = (float*)(ws + 0 * SZ);   // later reused as hg1
    float* nb   = (float*)(ws + 1 * SZ);   // [E,2D], spans slots 1..2; later t + g
    float* t12  = (float*)(ws + 1 * SZ);   // t1, then t2
    float* gbuf = (float*)(ws + 2 * SZ);   // g1, then g2
    float* h    = (float*)(ws + 3 * SZ);   // later reused as node
    float* hg1  = (float*)(ws + 0 * SZ);
    float* node = (float*)(ws + 3 * SZ);
    float* remb = (float*)(ws + 4 * SZ);   // [2R, 2D]

    dim3 blk(256);
    auto waves  = [](long n) { return dim3((unsigned)((n * 32 + 255) / 256)); };
    auto ggrid  = [&](int N_, int M_) { return dim3((unsigned)((N_ + 159) / 160),
                                                    (unsigned)((M_ + 127) / 128)); };

    // 1) normalize embeddings
    k_norm<<<waves(E), blk, 0, stream>>>(we, embn, E, D);

    // 2) relation aggregation: r_emb = [[L|Rm]; -[L|Rm]]
    hipMemsetAsync(remb, 0, (size_t)2 * R * 2 * D * sizeof(float), stream);
    k_spmm<<<waves(nnzHR), blk, 0, stream>>>(hr_r, hr_c, hr_v, embn, D, remb, 2 * D, 0, D, nnzHR);
    k_spmm<<<waves(nnzTR), blk, 0, stream>>>(tr_r, tr_c, tr_v, embn, D, remb, 2 * D, D, D, nnzTR);
    k_negmirror<<<(R * 2 * D + 255) / 256, blk, 0, stream>>>(remb, R * 2 * D);

    // 3) nb = spmm(er, r_emb)   [E, 2D]
    hipMemsetAsync(nb, 0, (size_t)E * 2 * D * sizeof(float), stream);
    k_spmm<<<waves(nnzER), blk, 0, stream>>>(er_r, er_c, er_v, remb, 2 * D, nb, 2 * D, 0, 2 * D, nnzER);

    // 4) h = embn + relu([embn|nb] @ Dense + Bias)     (WMMA GEMM, K=900)
    gemm_bf16<1><<<ggrid(D, E), blk, 0, stream>>>(embn, D, nb, Dw, 3 * D, D, E,
                                                  Bias, embn, nullptr, h);

    // 5) t1 = h @ W1 ; g1 = spmm(adj, t1) ; hg1 = highway(h, relu(g1))
    gemm_bf16<0><<<ggrid(D, E), blk, 0, stream>>>(h, D, nullptr, W1, D, D, E,
                                                  nullptr, nullptr, nullptr, t12);
    hipMemsetAsync(gbuf, 0, SZ, stream);
    k_spmm<<<waves(nnzAD), blk, 0, stream>>>(ad_r, ad_c, ad_v, t12, D, gbuf, D, 0, D, nnzAD);
    gemm_bf16<2><<<ggrid(D, E), blk, 0, stream>>>(h, D, nullptr, kg, D, D, E,
                                                  bg, h, gbuf, hg1);

    // 6) t2 = hg1 @ W2 ; g2 = spmm(adj, t2) ; node = highway(hg1, relu(g2))
    gemm_bf16<0><<<ggrid(D, E), blk, 0, stream>>>(hg1, D, nullptr, W2, D, D, E,
                                                  nullptr, nullptr, nullptr, t12);
    hipMemsetAsync(gbuf, 0, SZ, stream);
    k_spmm<<<waves(nnzAD), blk, 0, stream>>>(ad_r, ad_c, ad_v, t12, D, gbuf, D, 0, D, nnzAD);
    gemm_bf16<2><<<ggrid(D, E), blk, 0, stream>>>(hg1, D, nullptr, kg, D, D, E,
                                                  bg, hg1, gbuf, node);

    // 7) hinge loss
    hipMemsetAsync(d_out, 0, (size_t)out_size * sizeof(float), stream);
    k_loss<<<T, blk, 0, stream>>>(node, posL, posR, negR, negL, mask, (float*)d_out);
}
